// Model_15418932592810
// MI455X (gfx1250) — matrix operand order repaired and checked by reading, not compiled, not run on hardware
//
#include <hip/hip_runtime.h>
#include <hip/hip_bf16.h>
#include <math.h>

// ---------------------------------------------------------------------------
// Problem constants
// ---------------------------------------------------------------------------
#define BB   8
#define LL   512
#define DD   512
#define HH   8
#define DHH  64
#define DFFC 2048
#define NTOK (BB * LL)      // 4096 token rows
#define ZB   (BB * HH)      // 64 attention batches

typedef __attribute__((ext_vector_type(16))) _Float16 v16h;
typedef __attribute__((ext_vector_type(8)))  _Float16 v8h;
typedef __attribute__((ext_vector_type(8)))  float    v8f;

// ---------------------------------------------------------------------------
// WMMA NT GEMM:  C[m,n] = alpha * sum_k A[m,k] * B[n,k]  (+ bias[n]) (+ReLU)
// Both operands are K-major rows (covers x @ W^T, Q @ K^T, P @ V via V^T).
// Batched over blockIdx.z with (b,h) stride decomposition.
// Block = 256 threads (8 waves) -> 128x64 C tile; wave -> 16x64 strip.
// B tile (64 x 32 f16) is shared by all 8 waves: staged through LDS with
// CDNA5 async global->LDS loads (ASYNCcnt), double-buffered so the next
// K-tile streams in while the current one feeds the WMMAs.
// ---------------------------------------------------------------------------
__global__ __launch_bounds__(256) void gemm_nt_k(
    const _Float16* __restrict__ A, long sAb, long sAh, int lda,
    const _Float16* __restrict__ Bm, long sBb, long sBh, int ldb,
    float* __restrict__ C, long sCb, long sCh, int ldc,
    _Float16* __restrict__ C16,
    const float* __restrict__ bias,
    int K, float alpha, int relu, int Hdiv)
{
    __shared__ _Float16 sB[2][64 * 32];   // two 4KB B-tile buffers

    int z = blockIdx.z;
    int b = z / Hdiv, h = z - b * Hdiv;
    A  += (long)b * sAb + (long)h * sAh;
    Bm += (long)b * sBb + (long)h * sBh;
    long coff = (long)b * sCb + (long)h * sCh;

    int tid  = threadIdx.x;
    int lane = tid & 31;
    int wave = tid >> 5;
    int m0 = blockIdx.y * 128 + wave * 16;
    int n0 = blockIdx.x * 64;

    int arow  = m0 + (lane & 15);
    int akoff = (lane & 16) ? 8 : 0;    // A frag: K chunks at +akoff, +akoff+16
    int bkoff = (lane & 16) ? 8 : 0;    // B frag: K chunks at +bkoff, +bkoff+16

    // ---- async B staging: thread t moves 16 bytes of the 64x32 tile ----
    int srow   = tid >> 2;              // 0..63  (row of B tile)
    int schunk = (tid & 3) * 8;         // f16 offset within 32-wide K slice
    const _Float16* gB = Bm + (long)(n0 + srow) * ldb + schunk;
    unsigned lds0 = (unsigned)(size_t)(&sB[0][srow * 32 + schunk]);
    unsigned lds1 = (unsigned)(size_t)(&sB[1][srow * 32 + schunk]);

    {   // prologue: stage tile for kk = 0 into buffer 0
        unsigned long long ga = (unsigned long long)(size_t)(gB);
        asm volatile("global_load_async_to_lds_b128 %0, %1, off"
                     :: "v"(lds0), "v"(ga) : "memory");
        asm volatile("s_wait_asynccnt 0x0" ::: "memory");
    }
    __syncthreads();

    v8f acc0 = {}, acc1 = {}, acc2 = {}, acc3 = {};
    int cur = 0;

    for (int kk = 0; kk < K; kk += 32) {
        // stage next K-tile into the other buffer while we compute
        if (kk + 32 < K) {
            unsigned long long ga = (unsigned long long)(size_t)(gB + (kk + 32));
            asm volatile("global_load_async_to_lds_b128 %0, %1, off"
                         :: "v"(cur ? lds0 : lds1), "v"(ga) : "memory");
        }

        // ---- A fragment (16x32 f16 per ISA layout) from global ----
        v16h af;
        {
            const _Float16* p = A + (long)arow * lda + kk + akoff;
            __builtin_prefetch(p + 32, 0, 1);
            v8h lo = *(const v8h*)p;
            v8h hi = *(const v8h*)(p + 16);
            #pragma unroll
            for (int i = 0; i < 8; ++i) { af[i] = lo[i]; af[i + 8] = hi[i]; }
        }
        // ---- 4 B fragments (32x16 f16 per ISA layout) from LDS ----
        const _Float16* bbase = &sB[cur][0];
        v16h bf[4];
        #pragma unroll
        for (int j = 0; j < 4; ++j) {
            const _Float16* p = bbase + (j * 16 + (lane & 15)) * 32 + bkoff;
            v8h lo = *(const v8h*)p;
            v8h hi = *(const v8h*)(p + 16);
            #pragma unroll
            for (int i = 0; i < 8; ++i) { bf[j][i] = lo[i]; bf[j][i + 8] = hi[i]; }
        }
        acc0 = __builtin_amdgcn_wmma_f32_16x16x32_f16(false, af, false, bf[0], (short)0, acc0, false, false);
        acc1 = __builtin_amdgcn_wmma_f32_16x16x32_f16(false, af, false, bf[1], (short)0, acc1, false, false);
        acc2 = __builtin_amdgcn_wmma_f32_16x16x32_f16(false, af, false, bf[2], (short)0, acc2, false, false);
        acc3 = __builtin_amdgcn_wmma_f32_16x16x32_f16(false, af, false, bf[3], (short)0, acc3, false, false);

        // wait for our async stores into the next buffer, then flip
        asm volatile("s_wait_asynccnt 0x0" ::: "memory");
        __syncthreads();
        cur ^= 1;
    }

    int ncol = lane & 15;
    int mrow = m0 + ((lane & 16) ? 8 : 0);
    #pragma unroll
    for (int j = 0; j < 4; ++j) {
        v8f acc = (j == 0) ? acc0 : (j == 1) ? acc1 : (j == 2) ? acc2 : acc3;
        int nc = n0 + j * 16 + ncol;
        float bv = bias ? bias[nc] : 0.f;
        #pragma unroll
        for (int v = 0; v < 8; ++v) {
            float val = alpha * acc[v] + bv;
            if (relu) val = fmaxf(val, 0.f);
            long o = coff + (long)(mrow + v) * ldc + nc;
            C[o] = val;
            if (C16) C16[o] = (_Float16)val;
        }
    }
}

// ---------------------------------------------------------------------------
// Elementwise / helper kernels
// ---------------------------------------------------------------------------
__global__ void cvt16_k(const float* __restrict__ X, _Float16* __restrict__ Y, long n) {
    long i = (long)blockIdx.x * 256 + threadIdx.x;
    if (i < n) Y[i] = (_Float16)X[i];
}

__global__ void ew_add_k(const float* __restrict__ A, const float* __restrict__ B,
                         float* __restrict__ Y, _Float16* __restrict__ Y16, long n) {
    long i = (long)blockIdx.x * 256 + threadIdx.x;
    if (i >= n) return;
    float v = A[i] + B[i];
    if (Y)   Y[i]   = v;
    if (Y16) Y16[i] = (_Float16)v;
}

// Reorganize f32 Q/K/V (token-major, head-interleaved) into z-major f16
// q16/k16: (z, L, 64); vT16: (z, 64, L)
__global__ void prep_qkv_k(const float* __restrict__ Q, int ldq,
                           const float* __restrict__ Kt, int ldk,
                           const float* __restrict__ V, int ldv,
                           _Float16* __restrict__ q16, _Float16* __restrict__ k16,
                           _Float16* __restrict__ vT16) {
    int d = threadIdx.x;            // 0..63
    int t = blockIdx.x;             // 0..511
    int z = blockIdx.y;             // 0..63
    int b = z / HH, h = z - b * HH;
    long row = (long)b * LL + t;
    long zb  = (long)z * LL * DHH;
    q16 [zb + (long)t * DHH + d] = (_Float16)Q [row * ldq + h * DHH + d];
    k16 [zb + (long)t * DHH + d] = (_Float16)Kt[row * ldk + h * DHH + d];
    vT16[zb + (long)d * LL  + t] = (_Float16)V [row * ldv + h * DHH + d];
}

// Softmax over last dim (Lk=512) with on-the-fly bias.
// mode 0: none; mode 1: damped decay (p0=lam_f, p1=lam_b); mode 2: phase cos (p0=logw)
__global__ __launch_bounds__(256) void softmax_bias_k(
    const float* __restrict__ S, _Float16* __restrict__ P,
    int mode, const float* __restrict__ p0, const float* __restrict__ p1)
{
    int q = blockIdx.x, z = blockIdx.y, h = z % HH;
    const float* row = S + ((long)z * LL + q) * LL;
    _Float16*   prow = P + ((long)z * LL + q) * LL;
    int tid = threadIdx.x;

    float lf = 0.f, lb = 0.f, w = 0.f;
    if (mode == 1) { lf = log1pf(expf(p0[h])); lb = log1pf(expf(p1[h])); }
    else if (mode == 2) { w = expf(p0[h]); }

    float v[2];
    #pragma unroll
    for (int i = 0; i < 2; ++i) {
        int k = tid + i * 256;
        float x = row[k];
        if (mode == 1) {
            float rel = (float)(k - q);                 // rel = k - q
            x += (rel > 0.f) ? (-lf * rel) : (lb * rel); // rel<0: -lb*|rel| = lb*rel
        } else if (mode == 2) {
            x += cosf(6.28318530717958647692f * w * (float)(q - k));
        }
        v[i] = x;
    }
    __shared__ float red[256];
    red[tid] = fmaxf(v[0], v[1]); __syncthreads();
    for (int s = 128; s > 0; s >>= 1) { if (tid < s) red[tid] = fmaxf(red[tid], red[tid + s]); __syncthreads(); }
    float mx = red[0]; __syncthreads();
    float e0 = expf(v[0] - mx), e1 = expf(v[1] - mx);
    red[tid] = e0 + e1; __syncthreads();
    for (int s = 128; s > 0; s >>= 1) { if (tid < s) red[tid] += red[tid + s]; __syncthreads(); }
    float inv = 1.0f / red[0];
    prow[tid]       = (_Float16)(e0 * inv);
    prow[tid + 256] = (_Float16)(e1 * inv);
}

// LayerNorm over D=512 per row
__global__ __launch_bounds__(256) void layernorm_k(const float* __restrict__ X,
    const float* __restrict__ g, const float* __restrict__ bt, float* __restrict__ Y)
{
    long row = blockIdx.x;
    const float* x = X + row * DD;
    int tid = threadIdx.x;
    float a0 = x[tid], a1 = x[tid + 256];
    __shared__ float red[256];
    red[tid] = a0 + a1; __syncthreads();
    for (int s = 128; s > 0; s >>= 1) { if (tid < s) red[tid] += red[tid + s]; __syncthreads(); }
    float mu = red[0] * (1.0f / DD); __syncthreads();
    float d0 = a0 - mu, d1 = a1 - mu;
    red[tid] = d0 * d0 + d1 * d1; __syncthreads();
    for (int s = 128; s > 0; s >>= 1) { if (tid < s) red[tid] += red[tid + s]; __syncthreads(); }
    float rstd = rsqrtf(red[0] * (1.0f / DD) + 1e-5f);
    Y[row * DD + tid]       = d0 * rstd * g[tid]       + bt[tid];
    Y[row * DD + tid + 256] = d1 * rstd * g[tid + 256] + bt[tid + 256];
}

// Per-batch per-feature mean over L (axis=1)
__global__ void col_mean_k(const float* __restrict__ X, float* __restrict__ Mb) {
    int idx = blockIdx.x * 256 + threadIdx.x;           // B*D
    int b = idx / DD, d = idx - b * DD;
    float s = 0.f;
    for (int t = 0; t < LL; ++t) s += X[((long)b * LL + t) * DD + d];
    Mb[idx] = s * (1.0f / LL);
}

__global__ void sub_colmean_k(const float* __restrict__ X, const float* __restrict__ Mb,
                              float* __restrict__ Y, _Float16* __restrict__ Y16) {
    long i = (long)blockIdx.x * 256 + threadIdx.x;      // B*L*D
    int d = (int)(i & (DD - 1));
    int b = (int)(i >> 18);                              // i / (L*D)
    float v = X[i] - Mb[b * DD + d];
    Y[i] = v;
    if (Y16) Y16[i] = (_Float16)v;
}

// series_decomp: 25-tap edge-clamped moving average
__global__ void decomp_k(const float* __restrict__ X,
                         float* __restrict__ Season, float* __restrict__ Trend) {
    long i = (long)blockIdx.x * 256 + threadIdx.x;      // B*L*D
    int d = (int)(i & (DD - 1));
    long bt = i >> 9;
    int t = (int)(bt & (LL - 1));
    int b = (int)(bt >> 9);
    const float* col = X + ((long)b * LL) * DD + d;
    float s = 0.f;
    #pragma unroll
    for (int j = -12; j <= 12; ++j) {
        int tt = t + j; tt = tt < 0 ? 0 : (tt > LL - 1 ? LL - 1 : tt);
        s += col[(long)tt * DD];
    }
    float m = s * (1.0f / 25.0f);
    Trend[i]  = m;
    Season[i] = X[i] - m;
}

// Build [sa_out | broadcast summary] in f16 (width 2D)
__global__ void concat_k(const float* __restrict__ SA, const float* __restrict__ SUM,
                         _Float16* __restrict__ OUT) {
    long i = (long)blockIdx.x * 256 + threadIdx.x;      // NTOK * 1024
    long r = i >> 10; int c = (int)(i & 1023);
    int b = (int)(r >> 9);
    float v = (c < DD) ? SA[r * DD + c] : SUM[b * DD + (c - DD)];
    OUT[i] = (_Float16)v;
}

// y[n] = sum_k x[k]*W[n,k] + bias[n]   (single-vector linear)
__global__ void vec_linear_k(const float* __restrict__ x, const float* __restrict__ W,
                             const float* __restrict__ bias, float* __restrict__ y,
                             int N, int K) {
    int n = blockIdx.x * 256 + threadIdx.x;
    if (n >= N) return;
    float s = bias ? bias[n] : 0.f;
    const float* wr = W + (long)n * K;
    for (int k = 0; k < K; ++k) s += x[k] * wr[k];
    y[n] = s;
}

// Y[m,n] = sum_k X[m,k]*W[n,k] + bias[n]  (tiny M, scalar)
__global__ void small_linear_k(const float* __restrict__ X, const float* __restrict__ W,
                               const float* __restrict__ bias, float* __restrict__ Y,
                               int M, int N, int K) {
    int idx = blockIdx.x * 256 + threadIdx.x;
    if (idx >= M * N) return;
    int m = idx / N, n = idx - m * N;
    float s = bias ? bias[n] : 0.f;
    const float* xr = X + (long)m * K;
    const float* wr = W + (long)n * K;
    for (int k = 0; k < K; ++k) s += xr[k] * wr[k];
    Y[idx] = s;
}

// Context pooling: one query (qp, 512) vs kp/vp (B*L, D), 8 heads, out HO (B, D)
__global__ __launch_bounds__(256) void ctx_attn_k(const float* __restrict__ qp,
    const float* __restrict__ KP, const float* __restrict__ VP, float* __restrict__ HO)
{
    __shared__ float lg[LL];
    __shared__ float red[256];
    int z = blockIdx.x; int b = z / HH, h = z - b * HH;
    int tid = threadIdx.x;
    const float* qh = qp + h * DHH;
    for (int t = tid; t < LL; t += 256) {
        const float* kr = KP + ((long)b * LL + t) * DD + h * DHH;
        float s = 0.f;
        #pragma unroll
        for (int d = 0; d < DHH; ++d) s += qh[d] * kr[d];
        lg[t] = s * 0.125f;
    }
    __syncthreads();
    red[tid] = fmaxf(lg[tid], lg[tid + 256]); __syncthreads();
    for (int s = 128; s > 0; s >>= 1) { if (tid < s) red[tid] = fmaxf(red[tid], red[tid + s]); __syncthreads(); }
    float mx = red[0]; __syncthreads();
    float e0 = expf(lg[tid] - mx), e1 = expf(lg[tid + 256] - mx);
    red[tid] = e0 + e1; __syncthreads();
    for (int s = 128; s > 0; s >>= 1) { if (tid < s) red[tid] += red[tid + s]; __syncthreads(); }
    float inv = 1.0f / red[0];
    __syncthreads();
    lg[tid] = e0 * inv; lg[tid + 256] = e1 * inv;
    __syncthreads();
    if (tid < DHH) {
        float acc = 0.f;
        for (int t = 0; t < LL; ++t) acc += lg[t] * VP[((long)b * LL + t) * DD + h * DHH + tid];
        HO[(long)b * DD + h * DHH + tid] = acc;
    }
}

// ---------------------------------------------------------------------------
// Host orchestration
// ---------------------------------------------------------------------------
extern "C" void kernel_launch(void* const* d_in, const int* in_sizes, int n_in,
                              void* d_out, int out_size, void* d_ws, size_t ws_size,
                              hipStream_t stream) {
    (void)in_sizes; (void)n_in; (void)out_size; (void)ws_size;

    // ---- inputs (setup_inputs order) ----
    const float* x_sa     = (const float*)d_in[0];
    const float* x_q1     = (const float*)d_in[1];
    const float* x_q2     = (const float*)d_in[2];
    const float* sa_qkv_w = (const float*)d_in[3];
    const float* sa_qkv_b = (const float*)d_in[4];
    const float* sa_out_w = (const float*)d_in[5];
    const float* sa_out_b = (const float*)d_in[6];
    const float* sa_lam_f = (const float*)d_in[7];
    const float* sa_lam_b = (const float*)d_in[8];
    const float* cf_q_w = (const float*)d_in[9];  const float* cf_q_b = (const float*)d_in[10];
    const float* cf_k_w = (const float*)d_in[11]; const float* cf_k_b = (const float*)d_in[12];
    const float* cf_v_w = (const float*)d_in[13]; const float* cf_v_b = (const float*)d_in[14];
    const float* cf_o_w = (const float*)d_in[15]; const float* cf_o_b = (const float*)d_in[16];
    const float* cf_logw = (const float*)d_in[17];
    const float* cr_q_w = (const float*)d_in[18]; const float* cr_q_b = (const float*)d_in[19];
    const float* cr_k_w = (const float*)d_in[20]; const float* cr_k_b = (const float*)d_in[21];
    const float* cr_v_w = (const float*)d_in[22]; const float* cr_v_b = (const float*)d_in[23];
    const float* cr_o_w = (const float*)d_in[24]; const float* cr_o_b = (const float*)d_in[25];
    const float* cr_logw = (const float*)d_in[26];
    const float* global_q  = (const float*)d_in[27];
    const float* mha_in_w  = (const float*)d_in[28];
    const float* mha_in_b  = (const float*)d_in[29];
    const float* mha_out_w = (const float*)d_in[30];
    const float* mha_out_b = (const float*)d_in[31];
    const float* fusion_w  = (const float*)d_in[32];
    const float* fusion_b  = (const float*)d_in[33];
    const float* conv1_w   = (const float*)d_in[34];
    const float* conv2_w   = (const float*)d_in[35];
    const float* norm3_g = (const float*)d_in[36]; const float* norm3_b = (const float*)d_in[37];
    const float* normf_g = (const float*)d_in[38]; const float* normf_b = (const float*)d_in[39];
    const float* trend_w = (const float*)d_in[40]; const float* trend_b = (const float*)d_in[41];

    // ---- workspace bump allocator ----
    char* wsb = (char*)d_ws;
    size_t off = 0;
    auto allocB = [&](size_t bytes) -> void* {
        void* p = wsb + off; off = (off + bytes + 255) & ~(size_t)255; return p;
    };
    auto a16 = [&](size_t n) -> _Float16* { return (_Float16*)allocB(n * 2); };
    auto a32 = [&](size_t n) -> float*    { return (float*)allocB(n * 4); };

    const long TD = (long)NTOK * DD;          // 2,097,152
    // f16 weights
    _Float16* w_qkv16  = a16(3 * DD * DD);
    _Float16* w_saout16= a16(DD * DD);
    _Float16* w_cfq16  = a16(DD * DD); _Float16* w_cfk16 = a16(DD * DD);
    _Float16* w_cfv16  = a16(DD * DD); _Float16* w_cfo16 = a16(DD * DD);
    _Float16* w_crq16  = a16(DD * DD); _Float16* w_crk16 = a16(DD * DD);
    _Float16* w_crv16  = a16(DD * DD); _Float16* w_cro16 = a16(DD * DD);
    _Float16* w_mha16  = a16(3 * DD * DD);
    _Float16* w_fus16  = a16(DD * 2 * DD);
    _Float16* w_c1_16  = a16((size_t)DFFC * DD);
    _Float16* w_c2_16  = a16((size_t)DD * DFFC);
    _Float16* w_tr16   = a16(DD * DD);
    // f16 activations
    _Float16* xsa16   = a16(TD);
    _Float16* xq1_16  = a16(TD);
    _Float16* xq2_16  = a16(TD);
    _Float16* q16     = a16(TD);
    _Float16* k16     = a16(TD);
    _Float16* vT16    = a16(TD);
    _Float16* attn16  = a16(TD);
    _Float16* enr16   = a16(TD);
    _Float16* ca1_16  = a16(TD);
    _Float16* cat16   = a16(2 * TD);
    _Float16* xn16    = a16(TD);
    _Float16* tsum16  = a16(TD);
    _Float16* probs16 = a16((size_t)ZB * LL * LL);   // 16.8M elems; reused as mid16
    _Float16* mid16   = probs16;                     // alias (probs dead by FFN time)
    // f32 buffers
    float* scores = a32((size_t)ZB * LL * LL);       // 16.8M; reused as mid32
    float* mid32  = scores;                          // alias
    float* big    = a32((size_t)NTOK * 3 * DD);      // qkv / separate q,k,v slices
    float* fA  = a32(TD);   // attention out / ffn out / general
    float* fSA = a32(TD);   // sa_out (long-lived)
    float* fC  = a32(TD);   // ca2 -> ca1 -> trend2
    float* fE  = a32(TD);   // enriched -> trend1
    float* f1  = a32(TD);
    float* f2  = a32(TD);
    float* f3  = a32(TD);
    float* qp       = a32(DD);
    float* head_out = a32((size_t)BB * DD);
    float* summary  = a32((size_t)BB * DD);
    float* mb       = a32((size_t)BB * DD);

    float* out_x     = (float*)d_out;
    float* out_trend = (float*)d_out + TD;

    // ---- helpers ----
    auto cvt = [&](const float* src, _Float16* dst, long n) {
        cvt16_k<<<dim3((unsigned)((n + 255) / 256)), dim3(256), 0, stream>>>(src, dst, n);
    };
    auto gemm = [&](const _Float16* A, long sAb, long sAh, int lda,
                    const _Float16* Bp, long sBb, long sBh, int ldb,
                    float* C, long sCb, long sCh, int ldc, _Float16* C16,
                    const float* bias, int M, int N, int K, float alpha,
                    int relu, int nz, int Hdiv) {
        dim3 g((unsigned)(N / 64), (unsigned)(M / 128), (unsigned)nz);
        gemm_nt_k<<<g, dim3(256), 0, stream>>>(A, sAb, sAh, lda, Bp, sBb, sBh, ldb,
                                               C, sCb, sCh, ldc, C16, bias, K, alpha, relu, Hdiv);
    };
    auto lin = [&](const _Float16* A, int lda, const _Float16* W, int ldw,
                   float* C, _Float16* C16, const float* bias, int N, int K, int relu) {
        gemm(A, 0, 0, lda, W, 0, 0, ldw, C, 0, 0, N, C16, bias, NTOK, N, K, 1.0f, relu, 1, 1);
    };
    const long zQK = (long)LL * DHH;             // 32768
    const long zSC = (long)LL * LL;              // 262144
    auto attention = [&](const float* Qf, int ldq, const float* Kf, int ldk,
                         const float* Vf, int ldv, int mode,
                         const float* p0, const float* p1,
                         float* attn_out_f32, _Float16* attn_out_f16) {
        prep_qkv_k<<<dim3(LL, ZB), dim3(DHH), 0, stream>>>(Qf, ldq, Kf, ldk, Vf, ldv, q16, k16, vT16);
        // logits = (1/8) Q K^T : per-z 512x512x64
        gemm(q16, HH * zQK, zQK, DHH, k16, HH * zQK, zQK, DHH,
             scores, HH * zSC, zSC, LL, nullptr, nullptr, LL, LL, DHH, 0.125f, 0, ZB, HH);
        softmax_bias_k<<<dim3(LL, ZB), dim3(256), 0, stream>>>(scores, probs16, mode, p0, p1);
        // out = P @ V  (via V^T, NT form): per-z 512x64x512, scatter into (b,t,D) + h*64
        gemm(probs16, HH * zSC, zSC, LL, vT16, HH * zQK, zQK, LL,
             attn_out_f32, (long)LL * DD, DHH, DD, attn_out_f16,
             nullptr, LL, DHH, LL, 1.0f, 0, ZB, HH);
    };

    // ---- convert inputs & weights to f16 ----
    cvt(x_sa, xsa16, TD); cvt(x_q1, xq1_16, TD); cvt(x_q2, xq2_16, TD);
    cvt(sa_qkv_w, w_qkv16, 3L * DD * DD);
    cvt(sa_out_w, w_saout16, (long)DD * DD);
    cvt(cf_q_w, w_cfq16, (long)DD * DD); cvt(cf_k_w, w_cfk16, (long)DD * DD);
    cvt(cf_v_w, w_cfv16, (long)DD * DD); cvt(cf_o_w, w_cfo16, (long)DD * DD);
    cvt(cr_q_w, w_crq16, (long)DD * DD); cvt(cr_k_w, w_crk16, (long)DD * DD);
    cvt(cr_v_w, w_crv16, (long)DD * DD); cvt(cr_o_w, w_cro16, (long)DD * DD);
    cvt(mha_in_w, w_mha16, 3L * DD * DD);
    cvt(fusion_w, w_fus16, 2L * DD * DD);
    cvt(conv1_w, w_c1_16, (long)DFFC * DD);
    cvt(conv2_w, w_c2_16, (long)DD * DFFC);
    cvt(trend_w, w_tr16, (long)DD * DD);

    // ================= damped self-attention on x_sa =================
    lin(xsa16, DD, w_qkv16, DD, big, nullptr, sa_qkv_b, 3 * DD, DD, 0);  // qkv (4096x1536)
    attention(big, 3 * DD, big + DD, 3 * DD, big + 2 * DD, 3 * DD,
              1, sa_lam_f, sa_lam_b, fA, attn16);
    lin(attn16, DD, w_saout16, DD, fSA, nullptr, sa_out_b, DD, DD, 0);   // sa_out

    // ================= phase cross-attn: q=x_q2, kv=x_q1 =================
    lin(xq2_16, DD, w_cfq16, DD, big,            nullptr, cf_q_b, DD, DD, 0);
    lin(xq1_16, DD, w_cfk16, DD, big + TD,       nullptr, cf_k_b, DD, DD, 0);
    lin(xq1_16, DD, w_cfv16, DD, big + 2 * TD,   nullptr, cf_v_b, DD, DD, 0);
    attention(big, DD, big + TD, DD, big + 2 * TD, DD, 2, cf_logw, nullptr, fA, attn16);
    lin(attn16, DD, w_cfo16, DD, fC, nullptr, cf_o_b, DD, DD, 0);        // ca2_out
    ew_add_k<<<dim3((unsigned)(TD / 256)), dim3(256), 0, stream>>>(x_q2, fC, fE, enr16, TD); // enriched

    // ================= phase cross-attn: q=x_q1, kv=enriched =================
    lin(xq1_16, DD, w_crq16, DD, big,          nullptr, cr_q_b, DD, DD, 0);
    lin(enr16,  DD, w_crk16, DD, big + TD,     nullptr, cr_k_b, DD, DD, 0);
    lin(enr16,  DD, w_crv16, DD, big + 2 * TD, nullptr, cr_v_b, DD, DD, 0);
    attention(big, DD, big + TD, DD, big + 2 * TD, DD, 2, cr_logw, nullptr, fA, attn16);
    lin(attn16, DD, w_cro16, DD, fC, ca1_16, cr_o_b, DD, DD, 0);         // ca1_out

    // ================= context pooling (tiny) =================
    vec_linear_k<<<dim3(2), dim3(256), 0, stream>>>(global_q, mha_in_w, mha_in_b, qp, DD, DD);
    lin(ca1_16, DD, w_mha16 + (size_t)DD * DD,     DD, f1, nullptr, mha_in_b + DD,     DD, DD, 0); // kp
    lin(ca1_16, DD, w_mha16 + (size_t)2 * DD * DD, DD, f2, nullptr, mha_in_b + 2 * DD, DD, DD, 0); // vp
    ctx_attn_k<<<dim3(ZB), dim3(256), 0, stream>>>(qp, f1, f2, head_out);
    small_linear_k<<<dim3((BB * DD + 255) / 256), dim3(256), 0, stream>>>(
        head_out, mha_out_w, mha_out_b, summary, BB, DD, DD);

    // ================= fusion + my_Layernorm(normf) =================
    concat_k<<<dim3((unsigned)(2 * TD / 256)), dim3(256), 0, stream>>>(fSA, summary, cat16);
    lin(cat16, 2 * DD, w_fus16, 2 * DD, f1, nullptr, fusion_b, DD, 2 * DD, 0);   // fused
    ew_add_k<<<dim3((unsigned)(TD / 256)), dim3(256), 0, stream>>>(x_sa, f1, f2, nullptr, TD);
    layernorm_k<<<dim3(NTOK), dim3(256), 0, stream>>>(f2, normf_g, normf_b, f3);
    col_mean_k<<<dim3((BB * DD) / 256), dim3(256), 0, stream>>>(f3, mb);
    sub_colmean_k<<<dim3((unsigned)(TD / 256)), dim3(256), 0, stream>>>(f3, mb, f1, nullptr);

    // ================= decomp 1 =================
    decomp_k<<<dim3((unsigned)(TD / 256)), dim3(256), 0, stream>>>(f1, f2, fE); // season->f2, trend1->fE

    // ================= my_Layernorm(norm3) + FFN =================
    layernorm_k<<<dim3(NTOK), dim3(256), 0, stream>>>(f2, norm3_g, norm3_b, f3);
    col_mean_k<<<dim3((BB * DD) / 256), dim3(256), 0, stream>>>(f3, mb);
    sub_colmean_k<<<dim3((unsigned)(TD / 256)), dim3(256), 0, stream>>>(f3, mb, f1, xn16);
    lin(xn16, DD, w_c1_16, DD, mid32, mid16, nullptr, DFFC, DD, 1);     // relu(xn W1^T)
    lin(mid16, DFFC, w_c2_16, DFFC, fA, nullptr, nullptr, DD, DFFC, 0); // ffn out
    ew_add_k<<<dim3((unsigned)(TD / 256)), dim3(256), 0, stream>>>(f2, fA, f3, nullptr, TD); // x + ffn

    // ================= decomp 2 + trend head =================
    decomp_k<<<dim3((unsigned)(TD / 256)), dim3(256), 0, stream>>>(f3, out_x, fC); // season->out, trend2->fC
    ew_add_k<<<dim3((unsigned)(TD / 256)), dim3(256), 0, stream>>>(fE, fC, f1, tsum16, TD);
    lin(tsum16, DD, w_tr16, DD, out_trend, nullptr, trend_b, DD, DD, 0);
}
